// Net_71554155151864
// MI455X (gfx1250) — matrix operand order repaired and checked by reading, not compiled, not run on hardware
//
#include <hip/hip_runtime.h>
#include <cstdint>

// ---------------------------------------------------------------------------
// CDNA5 (gfx1250) implementation.
// Precision: bf16 WMMA (v_wmma_f32_16x16x32_bf16) with f32 accumulate.
// ---------------------------------------------------------------------------

typedef __bf16 bf16x16 __attribute__((ext_vector_type(16)));
typedef float  f32x8   __attribute__((ext_vector_type(8)));

union Frag { bf16x16 v; uint4 q[2]; };

// ------------------------- packed-weight offsets (bf16 elements) ------------
constexpr long OFF_PRE0  = 0;
constexpr long OFF_PRE1  = OFF_PRE0  + 128L * 128;
constexpr long OFF_SCS   = OFF_PRE1  + 64L  * 64;
constexpr long OFF_SCV   = OFF_SCS   + 2048L * 128;
constexpr long OFF_SS    = OFF_SCV   + 1024L * 64;
constexpr long OFF_VS    = OFF_SS    + 384L * 192;
constexpr long OFF_SV    = OFF_VS    + 192L * 192;
constexpr long OFF_VV    = OFF_SV    + 384L * 64;
constexpr long OFF_F1    = OFF_VV    + 192L * 64;
constexpr long OFF_F2    = OFF_F1    + 128L * 64;
constexpr long OFF_F3    = OFF_F2    + 64L  * 64;
constexpr long OFF_POST0 = OFF_F3    + 64L  * 192;
constexpr long OFF_POST1 = OFF_POST0 + 128L * 128;
constexpr long WP_TOTAL  = OFF_POST1 + 64L  * 64;

// ------------------------------- LDS layout --------------------------------
constexpr int S_A    = 0;                 // 32*512 bf16  = 32768
constexpr int S_CS   = S_A   + 32768;     // 32*192 f32   = 24576
constexpr int S_SIN  = S_CS  + 24576;     // 32*384 bf16  = 24576
constexpr int S_VIN  = S_SIN + 24576;     // 3*32*192 bf16= 36864
constexpr int S_ES   = S_VIN + 36864;     // 32*128 f32   = 16384
constexpr int S_EV   = S_ES  + 16384;     // 3*32*64 f32  = 24576
constexpr int S_OH   = S_EV  + 24576;     // 32*16 f32    = 2048
constexpr int S_SH   = S_OH  + 2048;      // 32*4 f32     = 512
constexpr int S_W    = S_SH  + 512;       // 32*192 f32   = 24576
constexpr int S_SCS  = S_W   + 24576;     // 32*128 f32   = 16384
constexpr int S_SCV  = S_SCS + 16384;     // 3*32*64 f32  = 24576
constexpr int S_ZS   = S_SCV + 24576;     // 32*128 f32   = 16384
constexpr int S_ZV   = S_ZS  + 16384;     // 3*32*64 f32  = 24576
constexpr int S_TV   = S_ZV  + 24576;     // 32*64 f32    = 8192
constexpr int S_GT   = S_TV  + 8192;      // 32*64 f32    = 8192
constexpr int S_BIN  = S_GT  + 8192;      // 16*4 f32     = 256
constexpr int S_IDX  = S_BIN + 256;       // 3*32 int     = 384
constexpr int SMEM_BYTES = S_IDX + 384;   // 285824 B  (<=320KB per WG, CDNA5)

// -------------------------------- helpers ----------------------------------
// A (16-bit, 16x32): lanes 0-15 hold row M=lane, K 0-7 & 16-23;
// lanes 16-31 hold K 8-15 & 24-31.  B packed as [ktile][n][k%32].
// C: vgpr r -> (M = r + 8*(lane>=16), N = lane&15).
__device__ __forceinline__ void gemm_bf16(const __bf16* __restrict__ A, int K,
                                          const __bf16* __restrict__ Bp, int N,
                                          float* __restrict__ Cs, bool accum,
                                          int lane, int wave) {
  const int ntiles = N >> 4;
  const int rn  = lane & 15;
  const int hi  = (lane >> 4) & 1;
  const int kbA = hi ? 8 : 0;
  const int kbB = hi ? 8 : 0;
  for (int t = wave; t < 2 * ntiles; t += 8) {
    const int mh = t & 1, nt = t >> 1;
    const __bf16* ap = A + (mh * 16 + rn) * K;
    const int n = nt * 16 + rn;
    f32x8 acc = {};
    for (int kk = 0; kk < K; kk += 32) {
      Frag a, b;
      a.q[0] = *(const uint4*)(ap + kk + kbA);
      a.q[1] = *(const uint4*)(ap + kk + kbA + 16);
      const __bf16* bp = Bp + ((long)(kk >> 5) * N + n) * 32 + kbB;
      b.q[0] = *(const uint4*)(bp);
      b.q[1] = *(const uint4*)(bp + 16);
      acc = __builtin_amdgcn_wmma_f32_16x16x32_bf16(
          false, a.v, false, b.v, (short)0, acc, false, false);
    }
    float* cp = Cs + (mh * 16 + hi * 8) * N + nt * 16 + rn;
    if (accum) { for (int r = 0; r < 8; ++r) cp[r * N] += acc[r]; }
    else       { for (int r = 0; r < 8; ++r) cp[r * N]  = acc[r]; }
  }
}

// ------------------------------ prep kernels -------------------------------
__global__ void pack_B(const float* __restrict__ W, __bf16* __restrict__ out,
                       int K, int N) {
  int idx = blockIdx.x * 256 + threadIdx.x;
  if (idx >= K * N) return;
  int k = idx / N, n = idx - k * N;
  out[((long)(k >> 5) * N + n) * 32 + (k & 31)] = (__bf16)W[idx];
}

__global__ void zero_stats(float* stats) {
  if (threadIdx.x < 64) stats[threadIdx.x] = 0.f;
}

// ------------------------------ main kernel --------------------------------
__global__ void __launch_bounds__(256)
edge_main(const float* __restrict__ node_fea, const float* __restrict__ oh_g,
          const float* __restrict__ sh_g,    const float* __restrict__ efea,
          const float* __restrict__ ell,     const int*   __restrict__ eidx,
          const int*   __restrict__ batch,
          const float* __restrict__ bpre0,   const float* __restrict__ bf1,
          const float* __restrict__ bf2,     const float* __restrict__ bf3,
          const float* __restrict__ bpost0,
          const __bf16* __restrict__ wp,     float* __restrict__ stats,
          float* __restrict__ out, int E) {
  extern __shared__ char smem[];
  __bf16* Abuf = (__bf16*)(smem + S_A);
  float*  Cs   = (float*) (smem + S_CS);
  __bf16* sinb = (__bf16*)(smem + S_SIN);
  __bf16* vin  = (__bf16*)(smem + S_VIN);
  float*  es   = (float*) (smem + S_ES);
  float*  ev   = (float*) (smem + S_EV);
  float*  oh   = (float*) (smem + S_OH);
  float*  shm  = (float*) (smem + S_SH);
  float*  wbuf = (float*) (smem + S_W);
  float*  scs  = (float*) (smem + S_SCS);
  float*  scv  = (float*) (smem + S_SCV);
  float*  zs   = (float*) (smem + S_ZS);
  float*  zv   = (float*) (smem + S_ZV);
  float*  tv   = (float*) (smem + S_TV);
  float*  gate = (float*) (smem + S_GT);
  float*  bins = (float*) (smem + S_BIN);
  int*    ie   = (int*)   (smem + S_IDX);
  int*    je   = ie + 32;
  int*    be   = je + 32;

  const int tid  = threadIdx.x;
  const int lane = tid & 31, wave = tid >> 5;
  const int e0   = blockIdx.x * 32;

  if (tid < 32) {
    int g = e0 + tid;
    int i = 0, j = 0;
    if (g < E) { i = eidx[g]; j = eidx[E + g]; }
    ie[tid] = i; je[tid] = j;
    be[tid] = (g < E) ? batch[i] : 0;
  }
  if (tid < 64) bins[tid] = 0.f;
  __syncthreads();

  // ---- stage edge data + node gathers ----
  for (int idx = tid; idx < 32 * 320; idx += 256) {
    int e = idx / 320, c = idx - e * 320, g = e0 + e;
    float f = (g < E) ? efea[(size_t)g * 320 + c] : 0.f;
    if (c < 128) es[e * 128 + c] = f;
    else { int r = c - 128; ev[((r % 3) * 32 + e) * 64 + (r / 3)] = f; }
  }
  for (int idx = tid; idx < 32 * 16; idx += 256) {
    int e = idx >> 4, g = e0 + e;
    oh[idx] = (g < E) ? oh_g[(size_t)g * 16 + (idx & 15)] : 0.f;
  }
  for (int idx = tid; idx < 32 * 4; idx += 256) {
    int e = idx >> 2, g = e0 + e;
    shm[idx] = (g < E) ? sh_g[(size_t)g * 4 + (idx & 3)] : 0.f;
  }
  for (int idx = tid; idx < 32 * 128; idx += 256) {
    int e = idx >> 7, u = idx & 127;
    sinb[e * 384 + u]       = (__bf16)node_fea[(size_t)ie[e] * 320 + u];
    sinb[e * 384 + 128 + u] = (__bf16)node_fea[(size_t)je[e] * 320 + u];
  }
  for (int idx = tid; idx < 3 * 32 * 64; idx += 256) {
    int m = idx / 2048, r = idx - m * 2048, e = r >> 6, v = r & 63;
    vin[(m * 32 + e) * 192 + v]      = (__bf16)node_fea[(size_t)ie[e] * 320 + 128 + v * 3 + m];
    vin[(m * 32 + e) * 192 + 64 + v] = (__bf16)node_fea[(size_t)je[e] * 320 + 128 + v * 3 + m];
  }
  __syncthreads();

  // ---- G1: s = es @ Wpre0 / sqrt(128) + bpre0 ----
  for (int idx = tid; idx < 32 * 128; idx += 256)
    Abuf[idx] = (__bf16)(es[idx] * 0.08838834765f);
  __syncthreads();
  gemm_bf16(Abuf, 128, wp + OFF_PRE0, 128, Cs, false, lane, wave);
  __syncthreads();
  for (int idx = tid; idx < 32 * 128; idx += 256) {
    int e = idx >> 7, u = idx & 127;
    sinb[e * 384 + 256 + u] = (__bf16)(Cs[idx] + bpre0[u]);
  }

  // ---- G2: v[m] = ev[m] @ Wpre1 / 8 ----
  for (int m = 0; m < 3; ++m) {
    __syncthreads();
    for (int idx = tid; idx < 32 * 64; idx += 256)
      Abuf[idx] = (__bf16)(ev[m * 2048 + idx] * 0.125f);
    __syncthreads();
    gemm_bf16(Abuf, 64, wp + OFF_PRE1, 64, Cs, false, lane, wave);
    __syncthreads();
    for (int idx = tid; idx < 32 * 64; idx += 256) {
      int e = idx >> 6, u = idx & 63;
      vin[(m * 32 + e) * 192 + 128 + u] = (__bf16)Cs[idx];
    }
  }

  // ---- G3: sc_s, K=2048 in 4 chunks of 512 (A[e,k]=es*oh, k=v*16+s) ----
  for (int c = 0; c < 4; ++c) {
    __syncthreads();
    for (int idx = tid; idx < 32 * 512; idx += 256) {
      int e = idx >> 9, kl = idx & 511;
      Abuf[idx] = (__bf16)(es[e * 128 + c * 32 + (kl >> 4)] *
                           oh[e * 16 + (kl & 15)] * 0.02209708691f);
    }
    __syncthreads();
    gemm_bf16(Abuf, 512, wp + OFF_SCS + (long)c * 512 * 128, 128, Cs, c != 0, lane, wave);
  }
  __syncthreads();
  for (int idx = tid; idx < 32 * 128; idx += 256) scs[idx] = Cs[idx];

  // ---- G4: sc_v per m, K=1024 in 2 chunks ----
  for (int m = 0; m < 3; ++m) {
    for (int c = 0; c < 2; ++c) {
      __syncthreads();
      for (int idx = tid; idx < 32 * 512; idx += 256) {
        int e = idx >> 9, kl = idx & 511;
        Abuf[idx] = (__bf16)(ev[m * 2048 + e * 64 + c * 32 + (kl >> 4)] *
                             oh[e * 16 + (kl & 15)] * 0.03125f);
      }
      __syncthreads();
      gemm_bf16(Abuf, 512, wp + OFF_SCV + (long)c * 512 * 64, 64, Cs, c != 0, lane, wave);
    }
    __syncthreads();
    for (int idx = tid; idx < 32 * 64; idx += 256) scv[m * 2048 + idx] = Cs[idx];
  }

  // ---- G5-G7: MLP on edge_length_embedded ----
  __syncthreads();
  for (int idx = tid; idx < 32 * 128; idx += 256) {
    int e = idx >> 7, g = e0 + e;
    Abuf[idx] = (__bf16)((g < E) ? ell[(size_t)g * 128 + (idx & 127)] : 0.f);
  }
  __syncthreads();
  gemm_bf16(Abuf, 128, wp + OFF_F1, 64, Cs, false, lane, wave);
  __syncthreads();
  for (int idx = tid; idx < 32 * 64; idx += 256) {
    float x = Cs[idx] + bf1[idx & 63];
    Abuf[idx] = (__bf16)(x / (1.f + __expf(-x)));
  }
  __syncthreads();
  gemm_bf16(Abuf, 64, wp + OFF_F2, 64, Cs, false, lane, wave);
  __syncthreads();
  for (int idx = tid; idx < 32 * 64; idx += 256) {
    float x = Cs[idx] + bf2[idx & 63];
    Abuf[idx] = (__bf16)(x / (1.f + __expf(-x)));
  }
  __syncthreads();
  gemm_bf16(Abuf, 64, wp + OFF_F3, 192, Cs, false, lane, wave);
  __syncthreads();
  for (int idx = tid; idx < 32 * 192; idx += 256)
    wbuf[idx] = Cs[idx] + bf3[idx % 192];

  // ---- G8+G9: out_s = (s_in*sh0)@Wss/sqrt(768) + (v_in.sh1)@Wvs/sqrt(1152) ----
  __syncthreads();
  for (int idx = tid; idx < 32 * 384; idx += 256) {
    int e = idx / 384;
    Abuf[idx] = (__bf16)((float)sinb[idx] * shm[e * 4] * 0.03608439182f);
  }
  __syncthreads();
  gemm_bf16(Abuf, 384, wp + OFF_SS, 192, Cs, false, lane, wave);
  __syncthreads();
  for (int idx = tid; idx < 32 * 192; idx += 256) {
    int e = idx / 192, k = idx - e * 192;
    float a = (float)vin[e * 192 + k]              * shm[e * 4 + 1]
            + (float)vin[(32 + e) * 192 + k]       * shm[e * 4 + 2]
            + (float)vin[(64 + e) * 192 + k]       * shm[e * 4 + 3];
    Abuf[idx] = (__bf16)(a * 0.02946278255f);
  }
  __syncthreads();
  gemm_bf16(Abuf, 192, wp + OFF_VS, 192, Cs, true, lane, wave);
  __syncthreads();
  for (int idx = tid; idx < 32 * 192; idx += 256) {
    int e = idx / 192, u = idx - e * 192;
    float o = Cs[idx];
    float sg = 1.f / (1.f + __expf(-o));
    if (u < 128) zs[e * 128 + u] = o * sg * wbuf[idx];           // silu * w_s
    else         gate[e * 64 + (u - 128)] = sg * wbuf[idx];      // sigmoid * w_v
  }

  // ---- G10: t = s_in @ Wsv / sqrt(768) ----
  __syncthreads();
  for (int idx = tid; idx < 32 * 384; idx += 256)
    Abuf[idx] = (__bf16)((float)sinb[idx] * 0.03608439182f);
  __syncthreads();
  gemm_bf16(Abuf, 384, wp + OFF_SV, 64, Cs, false, lane, wave);
  __syncthreads();
  for (int idx = tid; idx < 32 * 64; idx += 256) tv[idx] = Cs[idx];

  // ---- G11: z_v[m] = (t*sh1[m] + (v_in[m]*sh0)@Wvv/sqrt(384)) * gate ----
  for (int m = 0; m < 3; ++m) {
    __syncthreads();
    for (int idx = tid; idx < 32 * 192; idx += 256) {
      int e = idx / 192;
      Abuf[idx] = (__bf16)((float)vin[m * 32 * 192 + idx] * shm[e * 4] * 0.05103103631f);
    }
    __syncthreads();
    gemm_bf16(Abuf, 192, wp + OFF_VV, 64, Cs, false, lane, wave);
    __syncthreads();
    for (int idx = tid; idx < 32 * 64; idx += 256) {
      int e = idx >> 6;
      zv[m * 2048 + idx] = (tv[idx] * shm[e * 4 + 1 + m] + Cs[idx]) * gate[idx];
    }
  }

  // ---- G12: post0 ----
  __syncthreads();
  for (int idx = tid; idx < 32 * 128; idx += 256)
    Abuf[idx] = (__bf16)(zs[idx] * 0.08838834765f);
  __syncthreads();
  gemm_bf16(Abuf, 128, wp + OFF_POST0, 128, Cs, false, lane, wave);
  __syncthreads();
  for (int idx = tid; idx < 32 * 128; idx += 256)
    zs[idx] = Cs[idx] + bpost0[idx & 127] + scs[idx];

  // ---- G13: post1 per m ----
  for (int m = 0; m < 3; ++m) {
    __syncthreads();
    for (int idx = tid; idx < 32 * 64; idx += 256)
      Abuf[idx] = (__bf16)(zv[m * 2048 + idx] * 0.125f);
    __syncthreads();
    gemm_bf16(Abuf, 64, wp + OFF_POST1, 64, Cs, false, lane, wave);
    __syncthreads();
    for (int idx = tid; idx < 32 * 64; idx += 256)
      zv[m * 2048 + idx] = Cs[idx] + scv[m * 2048 + idx];
  }

  // ---- group stats + raw-z store ----
  __syncthreads();
  if (tid < 32) {
    int g = e0 + tid;
    if (g < E) {
      float s = 0.f, qs = 0.f;
      for (int u = 0; u < 128; ++u) { float z = zs[tid * 128 + u]; s += z; qs += z * z; }
      float qv = 0.f;
      for (int m = 0; m < 3; ++m)
        for (int u = 0; u < 64; ++u) { float z = zv[m * 2048 + tid * 64 + u]; qv += z * z; }
      int grp = be[tid];
      atomicAdd(&bins[grp * 4 + 0], s);
      atomicAdd(&bins[grp * 4 + 1], qs);
      atomicAdd(&bins[grp * 4 + 2], qv);
      atomicAdd(&bins[grp * 4 + 3], 1.f);
    }
  }
  for (int idx = tid; idx < 32 * 320; idx += 256) {
    int e = idx / 320, c = idx - e * 320, g = e0 + e;
    if (g < E) {
      float val;
      if (c < 128) val = zs[e * 128 + c];
      else { int r = c - 128; val = zv[(r % 3) * 2048 + e * 64 + (r / 3)]; }
      out[(size_t)g * 320 + c] = val;
    }
  }
  __syncthreads();
  if (tid < 64) atomicAdd(&stats[tid], bins[tid]);
}

// ---------------------------- finalize + normalize -------------------------
__global__ void finalize_stats(float* stats) {
  int g = threadIdx.x;
  if (g < 16) {
    float s = stats[g * 4 + 0], qs = stats[g * 4 + 1];
    float qv = stats[g * 4 + 2], c = fmaxf(stats[g * 4 + 3], 1.f);
    float mean = s / (c * 128.f);
    float var  = qs / (c * 128.f) - mean * mean;
    float* fin = stats + 64;
    fin[g * 3 + 0] = mean;
    fin[g * 3 + 1] = rsqrtf(fmaxf(var, 0.f) + 1e-5f);
    fin[g * 3 + 2] = rsqrtf(qv / (c * 192.f) + 1e-5f);
  }
}

__global__ void normalize_out(float* __restrict__ out,
                              const float* __restrict__ efea,
                              const int* __restrict__ eidx,
                              const int* __restrict__ batch,
                              const float* __restrict__ stats,
                              const float* __restrict__ gamma_s,
                              const float* __restrict__ beta_s,
                              const float* __restrict__ gamma_v, int E) {
  long idx = (long)blockIdx.x * 256 + threadIdx.x;
  if (idx >= (long)E * 320) return;
  int g = (int)(idx / 320), c = (int)(idx - (long)g * 320);
  int grp = batch[eidx[g]];
  const float* fin = stats + 64 + grp * 3;
  float z = out[idx], r;
  if (c < 128) r = efea[idx] + (z - fin[0]) * fin[1] * gamma_s[c] + beta_s[c];
  else { int v = (c - 128) / 3; r = efea[idx] + z * fin[2] * gamma_v[v]; }
  out[idx] = r;
}

// --------------------------------- launcher --------------------------------
extern "C" void kernel_launch(void* const* d_in, const int* in_sizes, int n_in,
                              void* d_out, int out_size, void* d_ws, size_t ws_size,
                              hipStream_t stream) {
  const float* node_fea = (const float*)d_in[0];
  const float* oh       = (const float*)d_in[1];
  const float* sh       = (const float*)d_in[2];
  const float* efea     = (const float*)d_in[3];
  const float* ell      = (const float*)d_in[4];
  const int*   eidx     = (const int*)  d_in[5];
  const int*   batch    = (const int*)  d_in[6];
  const float* Wsc_s    = (const float*)d_in[7];
  const float* Wsc_v    = (const float*)d_in[8];
  const float* Wpre0    = (const float*)d_in[9];
  const float* bpre0    = (const float*)d_in[10];
  const float* Wpre1    = (const float*)d_in[11];
  const float* Wss      = (const float*)d_in[12];
  const float* Wvs      = (const float*)d_in[13];
  const float* Wsv      = (const float*)d_in[14];
  const float* Wvv      = (const float*)d_in[15];
  const float* Wf1      = (const float*)d_in[16];
  const float* bf1      = (const float*)d_in[17];
  const float* Wf2      = (const float*)d_in[18];
  const float* bf2      = (const float*)d_in[19];
  const float* Wf3      = (const float*)d_in[20];
  const float* bf3      = (const float*)d_in[21];
  const float* Wpost0   = (const float*)d_in[22];
  const float* bpost0   = (const float*)d_in[23];
  const float* Wpost1   = (const float*)d_in[24];
  const float* gamma_s  = (const float*)d_in[25];
  const float* beta_s   = (const float*)d_in[26];
  const float* gamma_v  = (const float*)d_in[27];

  float*  out   = (float*)d_out;
  float*  stats = (float*)d_ws;                       // 64 acc + 48 finalized
  __bf16* wp    = (__bf16*)((char*)d_ws + 512);       // packed bf16 weights

  const int E = in_sizes[3] / 320;

  auto pk = [&](const float* W, int K, int N, long off) {
    int n = K * N;
    pack_B<<<(n + 255) / 256, 256, 0, stream>>>(W, wp + off, K, N);
  };
  pk(Wpre0,  128, 128, OFF_PRE0);
  pk(Wpre1,   64,  64, OFF_PRE1);
  pk(Wsc_s, 2048, 128, OFF_SCS);   // (v,s,u) row-major == (K=v*16+s, N=u)
  pk(Wsc_v, 1024,  64, OFF_SCV);
  pk(Wss,    384, 192, OFF_SS);
  pk(Wvs,    192, 192, OFF_VS);
  pk(Wsv,    384,  64, OFF_SV);
  pk(Wvv,    192,  64, OFF_VV);
  pk(Wf1,    128,  64, OFF_F1);
  pk(Wf2,     64,  64, OFF_F2);
  pk(Wf3,     64, 192, OFF_F3);
  pk(Wpost0, 128, 128, OFF_POST0);
  pk(Wpost1,  64,  64, OFF_POST1);

  zero_stats<<<1, 64, 0, stream>>>(stats);

  int nblk = (E + 31) / 32;
  edge_main<<<nblk, 256, SMEM_BYTES, stream>>>(
      node_fea, oh, sh, efea, ell, eidx, batch,
      bpre0, bf1, bf2, bf3, bpost0, wp, stats, out, E);

  finalize_stats<<<1, 32, 0, stream>>>(stats);

  long total = (long)E * 320;
  normalize_out<<<(int)((total + 255) / 256), 256, 0, stream>>>(
      out, efea, eidx, batch, stats, gamma_s, beta_s, gamma_v, E);
}
